// Qwen3NextFullAttention_PT_58686433132564
// MI455X (gfx1250) — matrix operand order repaired and checked by reading, not compiled, not run on hardware
//
#include <hip/hip_runtime.h>
#include <stdint.h>

// ---- types --------------------------------------------------------------
typedef __attribute__((ext_vector_type(8)))  __bf16 bh8;
typedef __attribute__((ext_vector_type(16))) __bf16 bh16;
typedef __attribute__((ext_vector_type(8)))  float  f32x8;

__device__ __forceinline__ uint16_t f2bf(float f) {
  uint32_t u = __builtin_bit_cast(uint32_t, f);
  return (uint16_t)((u + 0x7FFFu + ((u >> 16) & 1u)) >> 16);
}
__device__ __forceinline__ float bf2f(uint16_t h) {
  return __builtin_bit_cast(float, ((uint32_t)h) << 16);
}
__device__ __forceinline__ bh16 cat16(bh8 lo, bh8 hi) {
  return __builtin_shufflevector(lo, hi, 0,1,2,3,4,5,6,7,8,9,10,11,12,13,14,15);
}
__device__ __forceinline__ f32x8 wmma_bf(bh16 a, bh16 b, f32x8 c) {
  return __builtin_amdgcn_wmma_f32_16x16x32_bf16(false, a, false, b, (short)0, c, false, false);
}

// Problem constants
#define BB   4
#define SS   2048
#define HIDN 2048
#define NH   16
#define NKV  4
#define HD   128
#define MTOT (BB * SS)          // 8192
#define SCALE 0.08838834764831845f

// ---- prep kernels -------------------------------------------------------
__global__ __launch_bounds__(256) void cast_f32_bf16(const float* __restrict__ x,
                                                     uint16_t* __restrict__ y, int n) {
  int i = (blockIdx.x * 256 + threadIdx.x) * 4;
  if (i < n) {
    float4 v = *(const float4*)(x + i);
    ushort4 o;
    o.x = f2bf(v.x); o.y = f2bf(v.y); o.z = f2bf(v.z); o.w = f2bf(v.w);
    *(ushort4*)(y + i) = o;
  }
}

// src: R x C (f32, row-major) -> dst: C x R (bf16, row-major)
__global__ __launch_bounds__(256) void transpose_cast(const float* __restrict__ src,
                                                      uint16_t* __restrict__ dst,
                                                      int R, int C) {
  __shared__ float tile[32][33];
  int tx = threadIdx.x & 31, ty = threadIdx.x >> 5;
  int c0 = blockIdx.x * 32, r0 = blockIdx.y * 32;
#pragma unroll
  for (int i = 0; i < 32; i += 8)
    tile[ty + i][tx] = src[(size_t)(r0 + ty + i) * C + c0 + tx];
  __syncthreads();
#pragma unroll
  for (int i = 0; i < 32; i += 8)
    dst[(size_t)(c0 + ty + i) * R + r0 + tx] = f2bf(tile[tx][ty + i]);
}

// ---- tiled bf16 GEMM: C[M,N] = A[M,K] * BT[N,K]^T -----------------------
// 128x128 block tile, 8 waves (2x4), 64x32 per wave, double-buffered LDS.
enum { EP_BF16 = 0, EP_F32 = 1, EP_VSCAT = 2 };

template <int MODE>
__global__ __launch_bounds__(256) void gemm_bf16k(const uint16_t* __restrict__ A,
                                                  const uint16_t* __restrict__ BT,
                                                  void* __restrict__ Cout,
                                                  int M, int N, int K) {
  __shared__ __align__(16) uint16_t lA[2][128 * 40];  // pitch 40: conflict-free b128
  __shared__ __align__(16) uint16_t lB[2][128 * 40];
  const int t = threadIdx.x;
  const int w = t >> 5, lane = t & 31, l15 = lane & 15, hl = lane >> 4;
  const int wm = w >> 2, wn = w & 3;                 // 2 (M) x 4 (N) waves
  const int m0 = blockIdx.y * 128, n0 = blockIdx.x * 128;
  const int srow = t >> 1, shalf = t & 1;

  f32x8 acc[4][2];
#pragma unroll
  for (int mi = 0; mi < 4; ++mi)
#pragma unroll
    for (int ni = 0; ni < 2; ++ni)
#pragma unroll
      for (int r = 0; r < 8; ++r) acc[mi][ni][r] = 0.0f;

  const uint16_t* gA = A + (size_t)(m0 + srow) * K + shalf * 16;
  const uint16_t* gB = BT + (size_t)(n0 + srow) * K + shalf * 16;
  const int soff = srow * 40 + shalf * 16;

  // prologue: stage chunk 0
  uint4 ra0 = ((const uint4*)gA)[0], ra1 = ((const uint4*)gA)[1];
  uint4 rb0 = ((const uint4*)gB)[0], rb1 = ((const uint4*)gB)[1];
  { uint4* sa = (uint4*)(lA[0] + soff); sa[0] = ra0; sa[1] = ra1;
    uint4* sb = (uint4*)(lB[0] + soff); sb[0] = rb0; sb[1] = rb1; }
  __syncthreads();

  int buf = 0;
  for (int kt = 0; kt < K; kt += 32) {
    const bool more = (kt + 32) < K;
    if (more) {  // prefetch next chunk global -> regs (overlaps WMMAs below)
      const uint4* pa = (const uint4*)(gA + kt + 32);
      ra0 = pa[0]; ra1 = pa[1];
      const uint4* pb = (const uint4*)(gB + kt + 32);
      rb0 = pb[0]; rb1 = pb[1];
    }
    bh16 af[4], bf[2];
#pragma unroll
    for (int mi = 0; mi < 4; ++mi) {
      int row = wm * 64 + mi * 16 + l15;
      bh8 lo = *(const bh8*)(lA[buf] + row * 40 + 8 * hl);
      bh8 hi = *(const bh8*)(lA[buf] + row * 40 + 16 + 8 * hl);
      af[mi] = cat16(lo, hi);
    }
#pragma unroll
    for (int ni = 0; ni < 2; ++ni) {
      int row = wn * 32 + ni * 16 + l15;
      bh8 lo = *(const bh8*)(lB[buf] + row * 40 + 8 * hl);
      bh8 hi = *(const bh8*)(lB[buf] + row * 40 + 16 + 8 * hl);
      bf[ni] = cat16(lo, hi);
    }
#pragma unroll
    for (int mi = 0; mi < 4; ++mi)
#pragma unroll
      for (int ni = 0; ni < 2; ++ni)
        acc[mi][ni] = wmma_bf(af[mi], bf[ni], acc[mi][ni]);
    if (more) {  // store next chunk into the other buffer (no WAR hazard)
      uint4* sa = (uint4*)(lA[buf ^ 1] + soff); sa[0] = ra0; sa[1] = ra1;
      uint4* sb = (uint4*)(lB[buf ^ 1] + soff); sb[0] = rb0; sb[1] = rb1;
    }
    __syncthreads();  // single barrier per chunk
    buf ^= 1;
  }

#pragma unroll
  for (int mi = 0; mi < 4; ++mi)
#pragma unroll
    for (int ni = 0; ni < 2; ++ni)
#pragma unroll
      for (int r = 0; r < 8; ++r) {
        int row = m0 + wm * 64 + mi * 16 + r + 8 * hl;
        int col = n0 + wn * 32 + ni * 16 + l15;
        float v = acc[mi][ni][r];
        if (MODE == EP_BF16) {
          ((uint16_t*)Cout)[(size_t)row * N + col] = f2bf(v);
        } else if (MODE == EP_F32) {
          ((float*)Cout)[(size_t)row * N + col] = v;
        } else {  // V scatter -> vT[(b*NKV+kv)*HD + d][s]
          int b = row >> 11, s = row & (SS - 1);
          int kv = col >> 7, d = col & (HD - 1);
          ((uint16_t*)Cout)[((size_t)(b * NKV + kv) * HD + d) * SS + s] = f2bf(v);
        }
      }
}

// ---- RMS-norm + RoPE (one wave per (m, head)) ---------------------------
__global__ __launch_bounds__(256) void qnorm_rope(const uint16_t* __restrict__ qg,
                                                  const float* __restrict__ w,
                                                  const float* __restrict__ cosb,
                                                  const float* __restrict__ sinb,
                                                  uint16_t* __restrict__ qbf) {
  int gw = blockIdx.x * 8 + (threadIdx.x >> 5);
  int lane = threadIdx.x & 31;
  int m = gw >> 4, h = gw & 15;
  int b = m >> 11, s = m & (SS - 1);
  const uint16_t* src = qg + (size_t)m * 4096 + h * 256;
  float x[4], ss = 0.0f;
#pragma unroll
  for (int i = 0; i < 4; ++i) { x[i] = bf2f(src[lane + 32 * i]); ss += x[i] * x[i]; }
#pragma unroll
  for (int o = 16; o >= 1; o >>= 1) ss += __shfl_xor(ss, o, 32);
  float inv = rsqrtf(ss * (1.0f / 128.0f) + 1e-6f);
#pragma unroll
  for (int i = 0; i < 4; ++i) x[i] = x[i] * inv * (1.0f + w[lane + 32 * i]);
  float partner = __shfl_xor(x[0], 16, 32);          // d<32 lives in x[0]
  float c = cosb[(size_t)m * 32 + lane];
  float sn = sinb[(size_t)m * 32 + lane];
  x[0] = (lane < 16) ? (x[0] * c - partner * sn) : (x[0] * c + partner * sn);
  uint16_t* dst = qbf + ((size_t)(b * NH + h) * SS + s) * HD;
#pragma unroll
  for (int i = 0; i < 4; ++i) dst[lane + 32 * i] = f2bf(x[i]);
}

__global__ __launch_bounds__(256) void knorm_rope(const float* __restrict__ kf,
                                                  const float* __restrict__ w,
                                                  const float* __restrict__ cosb,
                                                  const float* __restrict__ sinb,
                                                  uint16_t* __restrict__ kbf) {
  int gw = blockIdx.x * 8 + (threadIdx.x >> 5);
  int lane = threadIdx.x & 31;
  int m = gw >> 2, kv = gw & 3;
  int b = m >> 11, s = m & (SS - 1);
  const float* src = kf + (size_t)m * 512 + kv * 128;
  float x[4], ss = 0.0f;
#pragma unroll
  for (int i = 0; i < 4; ++i) { x[i] = src[lane + 32 * i]; ss += x[i] * x[i]; }
#pragma unroll
  for (int o = 16; o >= 1; o >>= 1) ss += __shfl_xor(ss, o, 32);
  float inv = rsqrtf(ss * (1.0f / 128.0f) + 1e-6f);
#pragma unroll
  for (int i = 0; i < 4; ++i) x[i] = x[i] * inv * (1.0f + w[lane + 32 * i]);
  float partner = __shfl_xor(x[0], 16, 32);
  float c = cosb[(size_t)m * 32 + lane];
  float sn = sinb[(size_t)m * 32 + lane];
  x[0] = (lane < 16) ? (x[0] * c - partner * sn) : (x[0] * c + partner * sn);
  uint16_t* dst = kbf + ((size_t)(b * NKV + kv) * SS + s) * HD;
#pragma unroll
  for (int i = 0; i < 4; ++i) dst[lane + 32 * i] = f2bf(x[i]);
}

// ---- flash attention: one wave per 16-row query tile --------------------
__global__ __launch_bounds__(256) void attn_kernel(const uint16_t* __restrict__ qbf,
                                                   const uint16_t* __restrict__ kbf,
                                                   const uint16_t* __restrict__ vT,
                                                   const uint16_t* __restrict__ qg,
                                                   uint16_t* __restrict__ attnG) {
  __shared__ __align__(16) uint16_t lp[8][640];       // per-wave 16x32 bf16, pitch 40
  const int w = threadIdx.x >> 5, lane = threadIdx.x & 31;
  const int l15 = lane & 15, hl = lane >> 4;
  const int b = blockIdx.z, h = blockIdx.y;
  const int qbase = blockIdx.x * 128 + w * 16;
  const int kvh = h >> 2;                             // GQA group

  // Q fragments (A layout), held for the whole K loop
  const uint16_t* qrow = qbf + ((size_t)(b * NH + h) * SS + qbase + l15) * HD;
  bh16 qa[4];
#pragma unroll
  for (int c = 0; c < 4; ++c) {
    bh8 lo = *(const bh8*)(qrow + 32 * c + 8 * hl);
    bh8 hi = *(const bh8*)(qrow + 32 * c + 16 + 8 * hl);
    qa[c] = cat16(lo, hi);
  }
  const uint16_t* kb_base = kbf + (size_t)(b * NKV + kvh) * SS * HD;
  const uint16_t* vt_base = vT + (size_t)(b * NKV + kvh) * HD * SS;

  f32x8 acc[8];
  float mrow[8], lrow[8];
#pragma unroll
  for (int dt = 0; dt < 8; ++dt)
#pragma unroll
    for (int r = 0; r < 8; ++r) acc[dt][r] = 0.0f;
#pragma unroll
  for (int r = 0; r < 8; ++r) { mrow[r] = -3.0e38f; lrow[r] = 0.0f; }

  uint16_t* lpw = lp[w];
  const int qend = qbase + 16;
  for (int kb = 0; kb < qend; kb += 32) {
    // ---- scores S = Q K^T (two 16x16 tiles, K=128) ----
    f32x8 st[2];
#pragma unroll
    for (int tt = 0; tt < 2; ++tt) {
#pragma unroll
      for (int r = 0; r < 8; ++r) st[tt][r] = 0.0f;
      const uint16_t* krow = kb_base + (size_t)(kb + tt * 16 + l15) * HD;
#pragma unroll
      for (int c = 0; c < 4; ++c) {
        bh8 lo = *(const bh8*)(krow + 32 * c + 8 * hl);
        bh8 hi = *(const bh8*)(krow + 32 * c + 16 + 8 * hl);
        st[tt] = wmma_bf(qa[c], cat16(lo, hi), st[tt]);
      }
    }
    // ---- scale + causal mask + online softmax ----
#pragma unroll
    for (int r = 0; r < 8; ++r) {
      int row = qbase + r + 8 * hl;
      int c0 = kb + l15, c1 = kb + 16 + l15;
      float s0 = (c0 > row) ? -1.0e9f : st[0][r] * SCALE;
      float s1 = (c1 > row) ? -1.0e9f : st[1][r] * SCALE;
      float mx = fmaxf(s0, s1);
#pragma unroll
      for (int o = 8; o >= 1; o >>= 1) mx = fmaxf(mx, __shfl_xor(mx, o, 32));
      float mnew = fmaxf(mrow[r], mx);
      float alpha = __expf(mrow[r] - mnew);
      float p0 = __expf(s0 - mnew);
      float p1 = __expf(s1 - mnew);
      float ps = p0 + p1;
#pragma unroll
      for (int o = 8; o >= 1; o >>= 1) ps += __shfl_xor(ps, o, 32);
      lrow[r] = lrow[r] * alpha + ps;
      mrow[r] = mnew;
#pragma unroll
      for (int dt = 0; dt < 8; ++dt) acc[dt][r] *= alpha;
      lpw[(r + 8 * hl) * 40 + l15] = f2bf(p0);        // C-layout -> LDS
      lpw[(r + 8 * hl) * 40 + 16 + l15] = f2bf(p1);
    }
    asm volatile("s_wait_dscnt 0" ::: "memory");      // LDS RAW fence (same wave)
    // ---- P back as A-layout fragment ----
    bh8 plo = *(const bh8*)(lpw + l15 * 40 + 8 * hl);
    bh8 phi = *(const bh8*)(lpw + l15 * 40 + 16 + 8 * hl);
    bh16 pf = cat16(plo, phi);
    // ---- acc += P V (8 tiles over d) ----
#pragma unroll
    for (int dt = 0; dt < 8; ++dt) {
      const uint16_t* vrow = vt_base + (size_t)(16 * dt + l15) * SS + kb + 8 * hl;
      bh8 lo = *(const bh8*)(vrow);
      bh8 hi = *(const bh8*)(vrow + 16);
      acc[dt] = wmma_bf(pf, cat16(lo, hi), acc[dt]);
    }
  }
  // ---- epilogue: normalize, gate with sigmoid, write bf16 ----
#pragma unroll
  for (int dt = 0; dt < 8; ++dt)
#pragma unroll
    for (int r = 0; r < 8; ++r) {
      int row = qbase + r + 8 * hl;
      int d = 16 * dt + l15;
      float g = bf2f(qg[(size_t)(b * SS + row) * 4096 + h * 256 + 128 + d]);
      float sig = 1.0f / (1.0f + __expf(-g));
      float v = (acc[dt][r] / lrow[r]) * sig;
      attnG[(size_t)(b * SS + row) * 2048 + h * 128 + d] = f2bf(v);
    }
}

// ---- host side ----------------------------------------------------------
extern "C" void kernel_launch(void* const* d_in, const int* in_sizes, int n_in,
                              void* d_out, int out_size, void* d_ws, size_t ws_size,
                              hipStream_t stream) {
  (void)in_sizes; (void)n_in; (void)out_size; (void)ws_size;
  const float* hs   = (const float*)d_in[0];
  const float* cosb = (const float*)d_in[1];
  const float* sinb = (const float*)d_in[2];
  /* d_in[3] = attention_mask: causal, recomputed in-kernel */
  const float* wq = (const float*)d_in[4];
  const float* wk = (const float*)d_in[5];
  const float* wv = (const float*)d_in[6];
  const float* wo = (const float*)d_in[7];
  const float* qw = (const float*)d_in[8];
  const float* kw = (const float*)d_in[9];

  char* ws = (char*)d_ws;
  size_t off = 0;
  auto alloc = [&](size_t bytes) -> void* {
    void* p = ws + off;
    off += (bytes + 255) & ~(size_t)255;
    return p;
  };
  uint16_t* Xbf   = (uint16_t*)alloc((size_t)MTOT * HIDN * 2);        // 32 MB
  uint16_t* WqT   = (uint16_t*)alloc((size_t)4096 * HIDN * 2);        // 16 MB
  uint16_t* WkT   = (uint16_t*)alloc((size_t)512 * HIDN * 2);
  uint16_t* WvT   = (uint16_t*)alloc((size_t)512 * HIDN * 2);
  uint16_t* WoT   = (uint16_t*)alloc((size_t)2048 * 2048 * 2);        // 8 MB
  uint16_t* qg    = (uint16_t*)alloc((size_t)MTOT * 4096 * 2);        // 64 MB
  float*    kf    = (float*)   alloc((size_t)MTOT * 512 * 4);         // 16 MB
  uint16_t* vT    = (uint16_t*)alloc((size_t)BB * NKV * HD * SS * 2); // 8 MB
  uint16_t* qbf   = (uint16_t*)alloc((size_t)BB * NH * SS * HD * 2);  // 32 MB
  uint16_t* kbf   = (uint16_t*)alloc((size_t)BB * NKV * SS * HD * 2); // 8 MB
  uint16_t* attnG = (uint16_t*)alloc((size_t)MTOT * 2048 * 2);        // 32 MB

  // 1) precision conversion + weight transposes
  cast_f32_bf16<<<(MTOT * HIDN / 4 + 255) / 256, 256, 0, stream>>>(hs, Xbf, MTOT * HIDN);
  transpose_cast<<<dim3(4096 / 32, 2048 / 32), 256, 0, stream>>>(wq, WqT, 2048, 4096);
  transpose_cast<<<dim3(512 / 32, 2048 / 32), 256, 0, stream>>>(wk, WkT, 2048, 512);
  transpose_cast<<<dim3(512 / 32, 2048 / 32), 256, 0, stream>>>(wv, WvT, 2048, 512);
  transpose_cast<<<dim3(2048 / 32, 2048 / 32), 256, 0, stream>>>(wo, WoT, 2048, 2048);
  // 2) projections (bf16 WMMA, f32 accumulate, double-buffered LDS)
  gemm_bf16k<EP_BF16><<<dim3(4096 / 128, MTOT / 128), 256, 0, stream>>>(Xbf, WqT, qg, MTOT, 4096, HIDN);
  gemm_bf16k<EP_F32><<<dim3(512 / 128, MTOT / 128), 256, 0, stream>>>(Xbf, WkT, kf, MTOT, 512, HIDN);
  gemm_bf16k<EP_VSCAT><<<dim3(512 / 128, MTOT / 128), 256, 0, stream>>>(Xbf, WvT, vT, MTOT, 512, HIDN);
  // 3) RMS-norm + RoPE
  qnorm_rope<<<MTOT * NH / 8, 256, 0, stream>>>(qg, qw, cosb, sinb, qbf);
  knorm_rope<<<MTOT * NKV / 8, 256, 0, stream>>>(kf, kw, cosb, sinb, kbf);
  // 4) causal flash attention + sigmoid gating (K/V stay L2-resident: 16 MB << 192 MB)
  attn_kernel<<<dim3(SS / 128, NH, BB), 256, 0, stream>>>(qbf, kbf, vT, qg, attnG);
  // 5) output projection -> f32 d_out
  gemm_bf16k<EP_F32><<<dim3(2048 / 128, MTOT / 128), 256, 0, stream>>>(attnG, WoT, (float*)d_out, MTOT, 2048, 2048);
}
